// SwinV2_Unet_11879879540855
// MI455X (gfx1250) — matrix operand order repaired and checked by reading, not compiled, not run on hardware
//
#include <hip/hip_runtime.h>
#include <hip/hip_bf16.h>

typedef __attribute__((ext_vector_type(16))) _Float16 v16h;
typedef __attribute__((ext_vector_type(8)))  float    v8f;
typedef __attribute__((ext_vector_type(4)))  int      v4i;

#define DEVINL __device__ __forceinline__

#if defined(__has_builtin)
#if __has_builtin(__builtin_amdgcn_global_load_async_to_lds_b128)
#define HAVE_ASYNC_LDS 1
#endif
#endif
#ifndef HAVE_ASYNC_LDS
#define HAVE_ASYNC_LDS 0
#endif

#if HAVE_ASYNC_LDS
typedef __attribute__((address_space(1))) v4i* gas1p;
typedef __attribute__((address_space(3))) v4i* gas3p;
DEVINL void async_copy16(const void* g, void* l) {
  void* gv = const_cast<void*>(g);
  __builtin_amdgcn_global_load_async_to_lds_b128((gas1p)gv, (gas3p)l, 0, 0);
}
DEVINL void async_wait0() {
#if __has_builtin(__builtin_amdgcn_s_wait_asynccnt)
  __builtin_amdgcn_s_wait_asynccnt(0);
#else
  asm volatile("s_wait_asynccnt 0x0" ::: "memory");
#endif
}
#endif

DEVINL v8f wmma32(v16h a, v16h b, v8f c) {
  return __builtin_amdgcn_wmma_f32_16x16x32_f16(false, a, false, b, (short)0, c, false, false);
}

union FragV { v16h h; v4i q[2]; };

// A-fragment (16x32 f16, M x K): lane l16 = row M; VGPR v holds K pair.
// v<4: K0 = 2v + 8*half ; v>=4: K0 = 16 + 2(v-4) + 8*half   (ISA 7.12.2)
// dwords: base/2 + 4*half + {0..3}  and  base/2 + 8 + 4*half + {0..3}  (16B aligned)
DEVINL v16h afrag(const _Float16* p, int ld, int row, int kbase, int half) {
  FragV f;
  const unsigned* pu = (const unsigned*)p;
  int base = (row * ld + kbase) >> 1;
  f.q[0] = *(const v4i*)(pu + base + 4 * half);
  f.q[1] = *(const v4i*)(pu + base + 8 + 4 * half);
  return f.h;
}

// B-fragment (32x16 f16, K x N): lane l16 = col N; v<4: K0 = 2v + 8*half ; v>=4: K0 = 16 + 2(v-4) + 8*half.
// Memory layout: element (K,N) at p[col(N)*ld + K] -> two runs of 4 dwords, 8 dwords apart (16B aligned).
DEVINL v16h bfrag(const _Float16* p, int ld, int col, int kbase, int half) {
  FragV f;
  const unsigned* pu = (const unsigned*)p;
  int base = ((col * ld + kbase) >> 1) + 4 * half;
  f.q[0] = *(const v4i*)(pu + base);
  f.q[1] = *(const v4i*)(pu + base + 8);
  return f.h;
}

// ---------------- K1: Enformer positional features (511 x 768) ----------------
__global__ void k_posemb(float* __restrict__ pe) {
  int i = blockIdx.x;   // 0..510
  int j = threadIdx.x;  // 0..127  (nb = 128)
  float d  = (float)(i - 255);
  float ad = fabsf(d);
  float sgn = (d > 0.f) ? 1.f : ((d < 0.f) ? -1.f : 0.f);
  float fe = exp2f(-ad * exp2f(-(3.f + 5.f * (float)j / 127.f)));
  float cw = exp2f((float)(j + 1)) - 1.f;
  float fc = (cw > ad) ? 1.f : 0.f;
  double m = 2.0 + 2.0 * (double)j;   // mean (stddev = 1); conc = m^2, rate = m
  double c = m * m;
  float prob;
  if (ad == 0.f) {
    prob = 0.f;
  } else {
    double lp = (c - 1.0) * log((double)ad) - m * (double)ad - lgamma(c) + c * log(m);
    prob = (float)exp(lp);
  }
  prob += 1e-8f;
  __shared__ float red[128];
  red[j] = prob;
  __syncthreads();
  for (int s = 64; s > 0; s >>= 1) {
    if (j < s) red[j] = fmaxf(red[j], red[j + s]);
    __syncthreads();
  }
  float fg = prob / red[0];
  float* row = pe + (size_t)i * 768;
  row[j]       = fe;       row[128 + j] = fc;       row[256 + j] = fg;
  row[384 + j] = sgn * fe; row[512 + j] = sgn * fc; row[640 + j] = sgn * fg;
}

// ---------------- K2: rel_k = posemb @ w_rel, layout [h][jj(512)][d] f16 -------
__global__ void k_relk(const float* __restrict__ pe, const float* __restrict__ w_rel,
                       _Float16* __restrict__ relk) {
  int id = blockIdx.x * 256 + threadIdx.x;
  if (id >= 512 * 768) return;
  int jj = id / 768, n = id % 768;
  int h = n >> 6, d = n & 63;
  _Float16* out = relk + ((size_t)h * 512 + jj) * 64 + d;
  if (jj == 511) { *out = (_Float16)0.f; return; }  // zero pad row (shift guard)
  const float* prow = pe + (size_t)jj * 768;
  float acc = 0.f;
  for (int f = 0; f < 768; ++f) acc += prow[f] * w_rel[(size_t)f * 768 + n];
  *out = (_Float16)acc;
}

// ---------------- K3: f32 -> f16 convert ----------------
__global__ void k_cvt(const float* __restrict__ in, _Float16* __restrict__ out, int n) {
  int i = blockIdx.x * 256 + threadIdx.x;
  if (i < n) out[i] = (_Float16)in[i];
}

// ---------------- K4: QKV GEMM (16384x768 @ 768x2304) + bias/scatter ----------
// Block tile: 128(M) x 128(N), K-depth 64 per stage (12 stages).
__global__ __launch_bounds__(256) void k_qkv(const _Float16* __restrict__ xh,
                                             const _Float16* __restrict__ wh,
                                             const float* __restrict__ rcb,
                                             const float* __restrict__ rpb,
                                             _Float16* __restrict__ qcA,
                                             _Float16* __restrict__ qpA,
                                             _Float16* __restrict__ kA,
                                             _Float16* __restrict__ vA) {
  __shared__ __attribute__((aligned(16))) _Float16 As[128 * 64]; // [row][k] 16KB
  __shared__ __attribute__((aligned(16))) _Float16 Bs[128 * 64]; // [n][k]   16KB
  int t = threadIdx.x, wave = t >> 5, lane = t & 31, half = (lane >> 4) & 1, l16 = lane & 15;
  int Mbase = blockIdx.x * 128, Nbase = blockIdx.y * 128;
  v8f acc[8];
#pragma unroll
  for (int nc = 0; nc < 8; ++nc) acc[nc] = (v8f){0, 0, 0, 0, 0, 0, 0, 0};
  const unsigned* whu = (const unsigned*)wh;
#if !HAVE_ASYNC_LDS
  const unsigned* xu = (const unsigned*)xh;
#endif
  for (int kc = 0; kc < 12; ++kc) {
    int kb = kc * 64;
    // ---- A tile: 128 x 64 f16 ----
#if HAVE_ASYNC_LDS
#pragma unroll
    for (int q0 = 0; q0 < 4; ++q0) {
      int q = t + q0 * 256;          // 1024 16B chunks
      int row = q >> 2, seg = q & 3;
      async_copy16(xh + (size_t)(Mbase + row) * 768 + kb + seg * 8,
                   (char*)As + q * 16);
    }
#else
#pragma unroll
    for (int u0 = 0; u0 < 16; ++u0) {
      int u = t + u0 * 256;          // 4096 dwords
      int row = u >> 5, kp = u & 31;
      ((unsigned*)As)[u] = xu[(size_t)(Mbase + row) * 384 + (kb >> 1) + kp];
    }
#endif
    // ---- B tile: 64(k) x 128(n), staged transposed as [n][k] ----
#pragma unroll
    for (int u0 = 0; u0 < 16; ++u0) {
      int u = t + u0 * 256;          // 4096 dwords (pairs along n)
      int n2 = (u & 63) * 2, k = u >> 6;
      union { unsigned w; _Float16 h[2]; } cv;
      cv.w = whu[((size_t)(kb + k) * 2304 + Nbase + n2) >> 1];
      Bs[n2 * 64 + k]       = cv.h[0];
      Bs[(n2 + 1) * 64 + k] = cv.h[1];
    }
    // prefetch next K-chunk while this one is consumed
    if (kc + 1 < 12) {
      int kbn = kb + 64;
      __builtin_prefetch(xh + (size_t)(Mbase + (t & 127)) * 768 + kbn, 0, 3);
      __builtin_prefetch(wh + (size_t)(kbn + (t & 63)) * 2304 + Nbase + ((t >> 6) & 1) * 64, 0, 3);
    }
#if HAVE_ASYNC_LDS
    async_wait0();
#endif
    __syncthreads();
#pragma unroll
    for (int kcc = 0; kcc < 2; ++kcc) {
      v16h a = afrag(As, 64, wave * 16 + l16, kcc * 32, half);
#pragma unroll
      for (int nc = 0; nc < 8; ++nc) {
        v16h b = bfrag(Bs, 64, nc * 16 + l16, kcc * 32, half);
        acc[nc] = wmma32(a, b, acc[nc]);
      }
    }
    __syncthreads();
  }
#pragma unroll
  for (int nc = 0; nc < 8; ++nc)
#pragma unroll
    for (int r = 0; r < 8; ++r) {
      int n_g = Nbase + nc * 16 + l16;
      int row_g = Mbase + wave * 16 + r + 8 * half;
      float v = acc[nc][r];
      int h = n_g / 192, rem = n_g % 192, d = rem / 3, s = rem % 3;
      int bb = row_g >> 13, nn = row_g & 8191, wi = nn >> 8, ti = nn & 255;
      size_t off = ((((size_t)bb * 12 + h) * 32 + wi) * 256 + ti) * 64 + d;
      if (s == 0) {
        qcA[off] = (_Float16)(v + rcb[h * 64 + d]);
        qpA[off] = (_Float16)(v + rpb[h * 64 + d]);
      } else if (s == 1) {
        kA[off] = (_Float16)v;
      } else {
        vA[off] = (_Float16)v;
      }
    }
}

// ---------------- K5: fused windowed attention with relative shift ------------
__global__ __launch_bounds__(256) void k_attn(const _Float16* __restrict__ qcA,
                                              const _Float16* __restrict__ qpA,
                                              const _Float16* __restrict__ kA,
                                              const _Float16* __restrict__ vA,
                                              const _Float16* __restrict__ relk,
                                              _Float16* __restrict__ outh) {
  __shared__ __attribute__((aligned(16))) _Float16 Vt[64 * 256];   // 32 KB, [d][j]
  __shared__ __attribute__((aligned(16))) float    Gbuf[8 * 512];  // 16 KB
  __shared__ __attribute__((aligned(16))) _Float16 Pbuf[8 * 512];  // 8 KB
  int blk = blockIdx.x;                 // ((b*12)+h)*32 + wi
  int wi = blk & 31;
  int h  = (blk >> 5) % 12;
  int b  = blk / 384;
  size_t base = (size_t)blk * (256 * 64);
  const _Float16* qc = qcA + base;
  const _Float16* qp = qpA + base;
  const _Float16* kk = kA + base;
  const _Float16* vv = vA + base;
  const _Float16* rk = relk + (size_t)h * (512 * 64);
  int t = threadIdx.x, wave = t >> 5, lane = t & 31, half = (lane >> 4) & 1, l16 = lane & 15;
  float*    G = Gbuf + wave * 512;
  _Float16* P = Pbuf + wave * 512;

  // stage V transposed: read f16 pairs (d, d+1) as dwords, scatter rows
  const unsigned* vvu = (const unsigned*)vv;
  for (int idx = t; idx < 8192; idx += 256) {
    int j = idx >> 5, d2 = (idx & 31) * 2;
    union { unsigned w; _Float16 h2[2]; } cv;
    cv.w = vvu[idx];
    Vt[d2 * 256 + j]       = cv.h2[0];
    Vt[(d2 + 1) * 256 + j] = cv.h2[1];
  }
  __syncthreads();

  v16h aqc[2][2], aqp[2][2];
#pragma unroll
  for (int ii = 0; ii < 2; ++ii) {
    int row = (wave * 2 + ii) * 16 + l16;
#pragma unroll
    for (int kc = 0; kc < 2; ++kc) {
      aqc[ii][kc] = afrag(qc, 64, row, kc * 32, half);
      aqp[ii][kc] = afrag(qp, 64, row, kc * 32, half);
    }
  }

  float mro[2][8], lro[2][8];
#pragma unroll
  for (int ii = 0; ii < 2; ++ii)
#pragma unroll
    for (int r = 0; r < 8; ++r) { mro[ii][r] = -1e30f; lro[ii][r] = 0.f; }

  // ---- pass A: row max & sum(exp) ----
  for (int j16 = 0; j16 < 16; ++j16) {
    v16h bk0 = bfrag(kk, 64, j16 * 16 + l16, 0, half);
    v16h bk1 = bfrag(kk, 64, j16 * 16 + l16, 32, half);
#pragma unroll
    for (int ii = 0; ii < 2; ++ii) {
      int I = wave * 2 + ii;
      v8f c = (v8f){0, 0, 0, 0, 0, 0, 0, 0};
      c = wmma32(aqc[ii][0], bk0, c);
      c = wmma32(aqc[ii][1], bk1, c);
      int jjb = 240 + 16 * (j16 - I);   // band base: jj = 255 + (j - i)
#pragma unroll
      for (int nc = 0; nc < 2; ++nc) {
        v8f g = (v8f){0, 0, 0, 0, 0, 0, 0, 0};
        g = wmma32(aqp[ii][0], bfrag(rk, 64, jjb + nc * 16 + l16, 0, half), g);
        g = wmma32(aqp[ii][1], bfrag(rk, 64, jjb + nc * 16 + l16, 32, half), g);
#pragma unroll
        for (int r = 0; r < 8; ++r) G[(r + 8 * half) * 32 + nc * 16 + l16] = g[r];
      }
      asm volatile("s_wait_dscnt 0x0" ::: "memory");
#pragma unroll
      for (int r = 0; r < 8; ++r) {
        int si = r + 8 * half;
        float s = c[r] * 0.125f + G[si * 32 + 15 + l16 - si];
        float mo = mro[ii][r];
        float mn = fmaxf(mo, s);
        lro[ii][r] = lro[ii][r] * expf(mo - mn) + expf(s - mn);
        mro[ii][r] = mn;
      }
      asm volatile("s_wait_dscnt 0x0" ::: "memory");
    }
  }
  // butterfly merge across the 16 lanes holding one row
#pragma unroll
  for (int ii = 0; ii < 2; ++ii)
#pragma unroll
    for (int r = 0; r < 8; ++r) {
      float m = mro[ii][r], l = lro[ii][r];
#pragma unroll
      for (int mask = 1; mask < 16; mask <<= 1) {
        float m2 = __shfl_xor(m, mask, 32);
        float l2 = __shfl_xor(l, mask, 32);
        float mn = fmaxf(m, m2);
        l = l * expf(m - mn) + l2 * expf(m2 - mn);
        m = mn;
      }
      mro[ii][r] = m;
      lro[ii][r] = l;
    }

  // ---- pass B: recompute S, P = exp(S - m), O += P @ V ----
  v8f oacc[2][4];
#pragma unroll
  for (int ii = 0; ii < 2; ++ii)
#pragma unroll
    for (int nc = 0; nc < 4; ++nc) oacc[ii][nc] = (v8f){0, 0, 0, 0, 0, 0, 0, 0};

  for (int jp = 0; jp < 8; ++jp) {
#pragma unroll
    for (int ii = 0; ii < 2; ++ii) {
      int I = wave * 2 + ii;
#pragma unroll
      for (int jh = 0; jh < 2; ++jh) {
        int j16 = jp * 2 + jh;
        v16h bk0 = bfrag(kk, 64, j16 * 16 + l16, 0, half);
        v16h bk1 = bfrag(kk, 64, j16 * 16 + l16, 32, half);
        v8f c = (v8f){0, 0, 0, 0, 0, 0, 0, 0};
        c = wmma32(aqc[ii][0], bk0, c);
        c = wmma32(aqc[ii][1], bk1, c);
        int jjb = 240 + 16 * (j16 - I);
#pragma unroll
        for (int nc = 0; nc < 2; ++nc) {
          v8f g = (v8f){0, 0, 0, 0, 0, 0, 0, 0};
          g = wmma32(aqp[ii][0], bfrag(rk, 64, jjb + nc * 16 + l16, 0, half), g);
          g = wmma32(aqp[ii][1], bfrag(rk, 64, jjb + nc * 16 + l16, 32, half), g);
#pragma unroll
          for (int r = 0; r < 8; ++r) G[(r + 8 * half) * 32 + nc * 16 + l16] = g[r];
        }
        asm volatile("s_wait_dscnt 0x0" ::: "memory");
#pragma unroll
        for (int r = 0; r < 8; ++r) {
          int si = r + 8 * half;
          float s = c[r] * 0.125f + G[si * 32 + 15 + l16 - si];
          P[si * 32 + jh * 16 + l16] = (_Float16)expf(s - mro[ii][r]);
        }
      }
      asm volatile("s_wait_dscnt 0x0" ::: "memory");
      v16h ap = afrag(P, 32, l16, 0, half);
#pragma unroll
      for (int nc = 0; nc < 4; ++nc) {
        v16h bv = bfrag(Vt, 256, nc * 16 + l16, jp * 32, half);
        oacc[ii][nc] = wmma32(ap, bv, oacc[ii][nc]);
      }
      asm volatile("s_wait_dscnt 0x0" ::: "memory");
    }
  }

  size_t orow0 = (size_t)b * 8192 + (size_t)wi * 256;
#pragma unroll
  for (int ii = 0; ii < 2; ++ii)
#pragma unroll
    for (int nc = 0; nc < 4; ++nc)
#pragma unroll
      for (int r = 0; r < 8; ++r) {
        int si = r + 8 * half;
        int row = (wave * 2 + ii) * 16 + si;
        float v = oacc[ii][nc][r] / lro[ii][r];
        outh[(orow0 + row) * 768 + h * 64 + nc * 16 + l16] = (_Float16)v;
      }
}

// ---------------- K6: output projection (16384x768 @ 768x768) + bias ----------
// Block tile: 128(M) x 128(N), K-depth 64 per stage (12 stages).
__global__ __launch_bounds__(256) void k_outproj(const _Float16* __restrict__ oh,
                                                 const _Float16* __restrict__ wh,
                                                 const float* __restrict__ bias,
                                                 float* __restrict__ out) {
  __shared__ __attribute__((aligned(16))) _Float16 As[128 * 64];
  __shared__ __attribute__((aligned(16))) _Float16 Bs[128 * 64];
  int t = threadIdx.x, wave = t >> 5, lane = t & 31, half = (lane >> 4) & 1, l16 = lane & 15;
  int Mbase = blockIdx.x * 128, Nbase = blockIdx.y * 128;
  v8f acc[8];
#pragma unroll
  for (int nc = 0; nc < 8; ++nc) acc[nc] = (v8f){0, 0, 0, 0, 0, 0, 0, 0};
  const unsigned* whu = (const unsigned*)wh;
#if !HAVE_ASYNC_LDS
  const unsigned* ou = (const unsigned*)oh;
#endif
  for (int kc = 0; kc < 12; ++kc) {
    int kb = kc * 64;
#if HAVE_ASYNC_LDS
#pragma unroll
    for (int q0 = 0; q0 < 4; ++q0) {
      int q = t + q0 * 256;
      int row = q >> 2, seg = q & 3;
      async_copy16(oh + (size_t)(Mbase + row) * 768 + kb + seg * 8,
                   (char*)As + q * 16);
    }
#else
#pragma unroll
    for (int u0 = 0; u0 < 16; ++u0) {
      int u = t + u0 * 256;
      int row = u >> 5, kp = u & 31;
      ((unsigned*)As)[u] = ou[(size_t)(Mbase + row) * 384 + (kb >> 1) + kp];
    }
#endif
#pragma unroll
    for (int u0 = 0; u0 < 16; ++u0) {
      int u = t + u0 * 256;
      int n2 = (u & 63) * 2, k = u >> 6;
      union { unsigned w; _Float16 h[2]; } cv;
      cv.w = whu[((size_t)(kb + k) * 768 + Nbase + n2) >> 1];
      Bs[n2 * 64 + k]       = cv.h[0];
      Bs[(n2 + 1) * 64 + k] = cv.h[1];
    }
    if (kc + 1 < 12) {
      int kbn = kb + 64;
      __builtin_prefetch(oh + (size_t)(Mbase + (t & 127)) * 768 + kbn, 0, 3);
      __builtin_prefetch(wh + (size_t)(kbn + (t & 63)) * 768 + Nbase + ((t >> 6) & 1) * 64, 0, 3);
    }
#if HAVE_ASYNC_LDS
    async_wait0();
#endif
    __syncthreads();
#pragma unroll
    for (int kcc = 0; kcc < 2; ++kcc) {
      v16h a = afrag(As, 64, wave * 16 + l16, kcc * 32, half);
#pragma unroll
      for (int nc = 0; nc < 8; ++nc) {
        v16h bfr = bfrag(Bs, 64, nc * 16 + l16, kcc * 32, half);
        acc[nc] = wmma32(a, bfr, acc[nc]);
      }
    }
    __syncthreads();
  }
#pragma unroll
  for (int nc = 0; nc < 8; ++nc)
#pragma unroll
    for (int r = 0; r < 8; ++r) {
      int n_g = Nbase + nc * 16 + l16;
      int row_g = Mbase + wave * 16 + r + 8 * half;
      out[(size_t)row_g * 768 + n_g] = acc[nc][r] + bias[n_g];
    }
}

extern "C" void kernel_launch(void* const* d_in, const int* in_sizes, int n_in,
                              void* d_out, int out_size, void* d_ws, size_t ws_size,
                              hipStream_t stream) {
  (void)in_sizes; (void)n_in; (void)out_size; (void)ws_size;
  const float* x     = (const float*)d_in[0];
  const float* w_qkv = (const float*)d_in[1];
  const float* w_out = (const float*)d_in[2];
  const float* b_out = (const float*)d_in[3];
  const float* w_rel = (const float*)d_in[4];
  const float* rcb   = (const float*)d_in[5];
  const float* rpb   = (const float*)d_in[6];
  float* out = (float*)d_out;

  char* w = (char*)d_ws;
  size_t off = 0;
  auto alloc = [&](size_t bytes) -> void* {
    void* p = w + off;
    off += (bytes + 255) & ~(size_t)255;
    return p;
  };
  const size_t QKVN = (size_t)2 * 12 * 32 * 256 * 64;       // 12.58M elems
  float*    pe  = (float*)alloc((size_t)511 * 768 * 4);
  _Float16* xh  = (_Float16*)alloc((size_t)16384 * 768 * 2);
  _Float16* wqh = (_Float16*)alloc((size_t)768 * 2304 * 2);
  _Float16* woh = (_Float16*)alloc((size_t)768 * 768 * 2);
  _Float16* rkh = (_Float16*)alloc((size_t)12 * 512 * 64 * 2);
  _Float16* qcA = (_Float16*)alloc(QKVN * 2);
  _Float16* qpA = (_Float16*)alloc(QKVN * 2);
  _Float16* kAb = (_Float16*)alloc(QKVN * 2);
  _Float16* vAb = (_Float16*)alloc(QKVN * 2);
  _Float16* oh  = (_Float16*)alloc((size_t)16384 * 768 * 2);

  k_posemb<<<511, 128, 0, stream>>>(pe);
  k_relk<<<(512 * 768) / 256, 256, 0, stream>>>(pe, w_rel, rkh);
  k_cvt<<<(16384 * 768 + 255) / 256, 256, 0, stream>>>(x, xh, 16384 * 768);
  k_cvt<<<(768 * 2304 + 255) / 256, 256, 0, stream>>>(w_qkv, wqh, 768 * 2304);
  k_cvt<<<(768 * 768 + 255) / 256, 256, 0, stream>>>(w_out, woh, 768 * 768);
  k_qkv<<<dim3(128, 18), 256, 0, stream>>>(xh, wqh, rcb, rpb, qcA, qpA, kAb, vAb);
  k_attn<<<768, 256, 0, stream>>>(qcA, qpA, kAb, vAb, rkh, oh);
  k_outproj<<<dim3(128, 6), 256, 0, stream>>>(oh, woh, b_out, out);
}
